// StackedGRU_231027_15006615734214
// MI455X (gfx1250) — matrix operand order repaired and checked by reading, not compiled, not run on hardware
//
#include <hip/hip_runtime.h>
#include <hip/hip_bf16.h>
#include <stdint.h>

// ---------------------------------------------------------------------------
// MI455X (gfx1250): bf16 WMMA everywhere, LDS-resident W_hh shards across 32
// persistent workgroups, global-barrier-synced GRU steps. Conv inputs carry 4
// zero-padded leading time rows (no divergence on causal taps); bulk GEMMs are
// register-blocked 4 N-tiles/wave with hoisted B-fragment loads so the
// scheduler can overlap global loads with v_wmma issue.
// ---------------------------------------------------------------------------

#define NWG   32
#define WGTH  256

typedef __attribute__((ext_vector_type(16))) __bf16 v16bf;
typedef __attribute__((ext_vector_type(8)))  float  v8f;

union FragU { v16bf v; uint4 q[2]; };

// problem dims
constexpr int B_ = 64, CKIN = 64, CCH = 256, CAFF = 512, HID = 512, CEMG = 64;
constexpr int TB = 50, TI = 25, NB = 39, T0 = 20, G3 = 1536, KE = 5, LTOT = 1000;
constexpr int TP = 54;  // padded window length: 4 zero rows + 50

// ---- bf16 workspace element offsets ----
constexpr size_t SZ_WTAP1 = (size_t)KE * CCH * CKIN;
constexpr size_t SZ_W2TAP = (size_t)KE * CAFF * CCH;
constexpr size_t OFF_WTAP1 = 0;
constexpr size_t OFF_W1PS  = OFF_WTAP1 + SZ_WTAP1;          // prefix sums PS_1..PS_5
constexpr size_t OFF_W2TAP = OFF_W1PS + SZ_WTAP1;
constexpr size_t OFF_WIH   = OFF_W2TAP + SZ_W2TAP;
constexpr size_t OFF_WHH   = OFF_WIH + (size_t)G3 * HID;
constexpr size_t OFF_WRDO  = OFF_WHH + (size_t)G3 * HID;
constexpr size_t OFF_WFB   = OFF_WRDO + (size_t)CEMG * HID;
constexpr size_t OFF_KWIN  = OFF_WFB + (size_t)CAFF * CEMG;  // [bin][b][54][64] padded
constexpr size_t SZ_KWIN   = (size_t)T0 * B_ * TP * CKIN;
constexpr size_t OFF_CHPRE = OFF_KWIN + SZ_KWIN;             // [bin][b][54][256] padded
constexpr size_t SZ_CHPRE  = (size_t)T0 * B_ * TP * CCH;
constexpr size_t OFF_AFF   = OFF_CHPRE + SZ_CHPRE;           // [bin][b][50][512]
constexpr size_t OFF_CHFB  = OFF_AFF + (size_t)T0 * B_ * TB * CAFF;  // [b][54][256]
constexpr size_t OFF_AFFFB = OFF_CHFB + (size_t)B_ * TP * CCH;       // [b][50][512]
constexpr size_t OFF_HBUF  = OFF_AFFFB + (size_t)B_ * TB * CAFF;     // double buffered
constexpr size_t OFF_EMG   = OFF_HBUF + (size_t)2 * B_ * HID;
constexpr size_t OFF_EMGP  = OFF_EMG + (size_t)B_ * CEMG;
constexpr size_t OFF_FBV   = OFF_EMGP + (size_t)B_ * CEMG;
constexpr size_t BF16_ELEMS = OFF_FBV + (size_t)B_ * CAFF;

// ---- f32 workspace element offsets (base after bf16 region) ----
constexpr size_t FOFF_UV  = 0;                                // [2][5][B][CCH]
constexpr size_t SZ_UV    = (size_t)2 * 5 * B_ * CCH;
constexpr size_t FOFF_GI  = FOFF_UV + SZ_UV;                  // [NWG][TB][B][48]
constexpr size_t SZ_GI    = (size_t)NWG * TB * B_ * 48;
constexpr size_t FOFF_BAR = FOFF_GI + SZ_GI;

// ---------------------------------------------------------------------------
// WMMA helpers (layouts per CDNA5 ISA §7.12.2, wave32)
// ---------------------------------------------------------------------------
__device__ __forceinline__ v8f wmma_bf16(v16bf a, v16bf b, v8f c) {
  return __builtin_amdgcn_wmma_f32_16x16x32_bf16(false, a, false, b, (short)0, c,
                                                 false, false);
}

// A fragment from row-major A[M][lda]: lane l holds row m=l&15;
// elems 0..7 = K(k0 + 8g ..), elems 8..15 = K(k0 + 16 + 8g ..), g = l>>4.
__device__ __forceinline__ v16bf load_a16(const __bf16* A, int lda, int m0, int k0,
                                          int lane) {
  int m = lane & 15, g = lane >> 4;
  const __bf16* p = A + (size_t)(m0 + m) * lda + k0 + 8 * g;
  FragU f;
  f.q[0] = *(const uint4*)p;
  f.q[1] = *(const uint4*)(p + 16);
  return f.v;
}

// B fragment: colptr = per-lane pointer to K=0 of this lane's column n=l&15;
// elems 0..7 = K(k0 + 8g ..), elems 8..15 = K(k0 + 16 + 8g ..). K contiguous in memory (column-of-K layout).
__device__ __forceinline__ v16bf load_b16(const __bf16* colptr, int k0, int lane) {
  const __bf16* p = colptr + k0 + 8 * (lane >> 4);
  FragU f;
  f.q[0] = *(const uint4*)p;
  f.q[1] = *(const uint4*)(p + 16);
  return f.v;
}

__device__ __forceinline__ float sigm(float x) { return 1.f / (1.f + __expf(-x)); }

__device__ __forceinline__ v8f vzero8() {
  v8f z = {0.f, 0.f, 0.f, 0.f, 0.f, 0.f, 0.f, 0.f};
  return z;
}

// monotonic global barrier across NWG resident workgroups
__device__ __forceinline__ void gbar(unsigned* bar, unsigned& gen) {
  __threadfence();
  __syncthreads();
  if (threadIdx.x == 0) {
    gen++;
    __hip_atomic_fetch_add(bar, 1u, __ATOMIC_ACQ_REL, __HIP_MEMORY_SCOPE_AGENT);
    while (__hip_atomic_load(bar, __ATOMIC_ACQUIRE, __HIP_MEMORY_SCOPE_AGENT) <
           gen * (unsigned)NWG) {
      __builtin_amdgcn_s_sleep(4);
    }
  }
  __syncthreads();
}

// ---------------------------------------------------------------------------
// K0: convert weights to bf16 WMMA layouts; build padded kin windows; zero the
// pad rows of the conv staging buffers.
// ---------------------------------------------------------------------------
__global__ __launch_bounds__(256) void k_prep(
    const float* We1, const float* We2, const float* Wih, const float* Whh,
    const float* Wrdo, const float* Wfb, const float* kin, __bf16* bws) {
  const size_t n_pad1 = (size_t)T0 * B_ * 4 * CCH;  // chpre pad rows
  const size_t n_pad2 = (size_t)B_ * 4 * CCH;       // chfb pad rows
  const size_t total = SZ_WTAP1 * 2 + SZ_W2TAP + (size_t)G3 * HID * 2 +
                       (size_t)CEMG * HID + (size_t)CAFF * CEMG + SZ_KWIN +
                       n_pad1 + n_pad2;
  for (size_t idx = (size_t)blockIdx.x * blockDim.x + threadIdx.x; idx < total;
       idx += (size_t)gridDim.x * blockDim.x) {
    size_t i = idx;
    if (i < SZ_WTAP1) {  // wtap1[k][co][ci] <- We1[co][ci][k]
      int k = (int)(i / (CCH * CKIN));
      int r = (int)(i % (CCH * CKIN));
      int co = r / CKIN, ci = r % CKIN;
      bws[OFF_WTAP1 + i] = (__bf16)We1[((size_t)co * CKIN + ci) * KE + k];
      continue;
    }
    i -= SZ_WTAP1;
    if (i < SZ_WTAP1) {  // w1ps[j][co][ci] = PS_{j+1} = sum_{k<=j} W_k
      int j = (int)(i / (CCH * CKIN));
      int r = (int)(i % (CCH * CKIN));
      int co = r / CKIN, ci = r % CKIN;
      float s = 0.f;
      for (int k = 0; k <= j; k++) s += We1[((size_t)co * CKIN + ci) * KE + k];
      bws[OFF_W1PS + i] = (__bf16)s;
      continue;
    }
    i -= SZ_WTAP1;
    if (i < SZ_W2TAP) {  // w2tap[k][co][ci] <- We2[co][ci][k]
      int k = (int)(i / (CAFF * CCH));
      int r = (int)(i % (CAFF * CCH));
      int co = r / CCH, ci = r % CCH;
      bws[OFF_W2TAP + i] = (__bf16)We2[((size_t)co * CCH + ci) * KE + k];
      continue;
    }
    i -= SZ_W2TAP;
    if (i < (size_t)G3 * HID) { bws[OFF_WIH + i] = (__bf16)Wih[i]; continue; }
    i -= (size_t)G3 * HID;
    if (i < (size_t)G3 * HID) { bws[OFF_WHH + i] = (__bf16)Whh[i]; continue; }
    i -= (size_t)G3 * HID;
    if (i < (size_t)CEMG * HID) {  // last tap of W_rdo
      bws[OFF_WRDO + i] = (__bf16)Wrdo[i * 3 + 2];
      continue;
    }
    i -= (size_t)CEMG * HID;
    if (i < (size_t)CAFF * CEMG) { bws[OFF_WFB + i] = (__bf16)Wfb[i]; continue; }
    i -= (size_t)CAFF * CEMG;
    if (i < SZ_KWIN) {  // kwin[bin][b][row 0..53][ci]; rows 0..3 are zeros
      int bin = (int)(i / ((size_t)B_ * TP * CKIN));
      int r = (int)(i % ((size_t)B_ * TP * CKIN));
      int b = r / (TP * CKIN);
      int r2 = r % (TP * CKIN);
      int row = r2 / CKIN, ci = r2 % CKIN;
      float v = 0.f;
      if (row >= 4)
        v = kin[((size_t)b * CKIN + ci) * LTOT + bin * TI + (row - 4)];
      bws[OFF_KWIN + i] = (__bf16)v;
      continue;
    }
    i -= SZ_KWIN;
    if (i < n_pad1) {  // zero pad rows of chpre
      int bin = (int)(i / ((size_t)B_ * 4 * CCH));
      int r = (int)(i % ((size_t)B_ * 4 * CCH));
      int b = r / (4 * CCH);
      int r2 = r % (4 * CCH);
      int row = r2 / CCH, c = r2 % CCH;
      bws[OFF_CHPRE + (((size_t)bin * B_ + b) * TP + row) * CCH + c] = (__bf16)0.f;
      continue;
    }
    i -= n_pad1;
    {  // zero pad rows of chfb
      int b = (int)(i / (4 * CCH));
      int r2 = (int)(i % (4 * CCH));
      int row = r2 / CCH, c = r2 % CCH;
      bws[OFF_CHFB + ((size_t)b * TP + row) * CCH + c] = (__bf16)0.f;
    }
  }
}

// ---------------------------------------------------------------------------
// K1: conv1 for bins 0..19: ch = relu(conv(kin)+b1) + ees
// jobs: bin(20) x coT(16) x nblk(50); wave = 4 N-tiles, A reused, B hoisted.
// ---------------------------------------------------------------------------
__global__ __launch_bounds__(256) void k_conv1(__bf16* bws, const float* b1,
                                               const float* ees, const int* wees) {
  const int lane = threadIdx.x & 31, wave = threadIdx.x >> 5;
  int job = blockIdx.x * 8 + wave;
  if (job >= T0 * 16 * 50) return;
  int bin = job / (16 * 50);
  int r = job % (16 * 50);
  int mT = r / 50, nb = r % 50;
  const int nn = lane & 15, lg = lane >> 4;
  const __bf16* kwin = bws + OFF_KWIN + (size_t)bin * B_ * TP * CKIN;
  const __bf16* wt = bws + OFF_WTAP1;
  int bj[4], ttj[4];
  const __bf16* cb[4];
#pragma unroll
  for (int j = 0; j < 4; j++) {
    int col = (nb * 4 + j) * 16 + nn;
    bj[j] = col / TB;
    ttj[j] = col % TB;
    cb[j] = kwin + ((size_t)bj[j] * TP + ttj[j]) * CKIN;  // tap k adds k*CKIN
  }
  v8f acc[4];
#pragma unroll
  for (int j = 0; j < 4; j++) acc[j] = vzero8();
#pragma unroll
  for (int k = 0; k < KE; k++) {
    const __bf16* Ak = wt + (size_t)k * CCH * CKIN;
#pragma unroll
    for (int kk = 0; kk < CKIN; kk += 32) {
      v16bf av = load_a16(Ak, CKIN, mT * 16, kk, lane);
      v16bf bv[4];
#pragma unroll
      for (int j = 0; j < 4; j++) bv[j] = load_b16(cb[j] + k * CKIN, kk, lane);
#pragma unroll
      for (int j = 0; j < 4; j++) acc[j] = wmma_bf16(av, bv[j], acc[j]);
    }
  }
  int eon = *wees;
#pragma unroll
  for (int j = 0; j < 4; j++) {
    int b = bj[j], tt = ttj[j];
    __bf16* chp =
        bws + OFF_CHPRE + (((size_t)bin * B_ + b) * TP + 4 + tt) * CCH;
#pragma unroll
    for (int i2 = 0; i2 < 8; i2++) {
      int co = mT * 16 + i2 + 8 * lg;
      float v = acc[j][i2] + b1[co];
      v = v > 0.f ? v : 0.f;
      if (eon) v += ees[((size_t)b * CCH + co) * LTOT + bin * TI + tt];
      chp[co] = (__bf16)v;
    }
  }
}

// ---------------------------------------------------------------------------
// K2: conv2 for bins 0..19: aff = relu(conv(ch)+b2); 4 N-tiles per wave.
// jobs: bin(20) x coT(32) x nblk(50)
// ---------------------------------------------------------------------------
__global__ __launch_bounds__(256) void k_conv2(__bf16* bws, const float* b2) {
  const int lane = threadIdx.x & 31, wave = threadIdx.x >> 5;
  int job = blockIdx.x * 8 + wave;
  if (job >= T0 * 32 * 50) return;
  int bin = job / (32 * 50);
  int r = job % (32 * 50);
  int mT = r / 50, nb = r % 50;
  const int nn = lane & 15, lg = lane >> 4;
  const __bf16* ch = bws + OFF_CHPRE + (size_t)bin * B_ * TP * CCH;
  const __bf16* w2 = bws + OFF_W2TAP;
  int bj[4], ttj[4];
  const __bf16* cb[4];
#pragma unroll
  for (int j = 0; j < 4; j++) {
    int col = (nb * 4 + j) * 16 + nn;
    bj[j] = col / TB;
    ttj[j] = col % TB;
    cb[j] = ch + ((size_t)bj[j] * TP + ttj[j]) * CCH;
  }
  v8f acc[4];
#pragma unroll
  for (int j = 0; j < 4; j++) acc[j] = vzero8();
  for (int k = 0; k < KE; k++) {
    const __bf16* Ak = w2 + (size_t)k * CAFF * CCH;
#pragma unroll
    for (int kk = 0; kk < CCH; kk += 32) {
      v16bf av = load_a16(Ak, CCH, mT * 16, kk, lane);
      v16bf bv[4];
#pragma unroll
      for (int j = 0; j < 4; j++) bv[j] = load_b16(cb[j] + k * CCH, kk, lane);
#pragma unroll
      for (int j = 0; j < 4; j++) acc[j] = wmma_bf16(av, bv[j], acc[j]);
    }
  }
#pragma unroll
  for (int j = 0; j < 4; j++) {
    int b = bj[j], tt = ttj[j];
    __bf16* dst = bws + OFF_AFF + (((size_t)bin * B_ + b) * TB + tt) * CAFF;
#pragma unroll
    for (int i2 = 0; i2 < 8; i2++) {
      int co = mT * 16 + i2 + 8 * lg;
      float v = acc[j][i2] + b2[co];
      dst[co] = (__bf16)(v > 0.f ? v : 0.f);
    }
  }
}

// ---------------------------------------------------------------------------
// K3: persistent GRU engine. 32 WGs x 256 threads; WG owns 16 hidden units.
// LDS: s_whh[48][512] bf16 (W_hh shard), s_h[64][512] bf16 (full h),
//      s_gh[48][64] f32, s_gifb[48][64] f32.  ~136 KB of 320 KB/WGP.
// ---------------------------------------------------------------------------
__global__ __launch_bounds__(WGTH) void k_recur(
    __bf16* bws, float* fws, float* out, const float* b1, const float* b2,
    const float* b_ih, const float* b_hh, const float* b_rdo, const float* b_fb,
    const float* ees, const int* wees) {
  extern __shared__ char smem[];
  __bf16* s_whh = (__bf16*)smem;               // 48*512
  __bf16* s_h = s_whh + 48 * HID;              // 64*512
  float* s_gh = (float*)(s_h + B_ * HID);      // 48*64
  float* s_gifb = s_gh + 48 * B_;              // 48*64

  const int tid = threadIdx.x, lane = tid & 31, wave = tid >> 5;
  const int wg = blockIdx.x;
  const int lg = lane >> 4, nn = lane & 15;
  unsigned gen = 0;
  unsigned* bar = (unsigned*)(fws + FOFF_BAR);
  float* giW = fws + FOFF_GI + (size_t)wg * TB * B_ * 48;
  float* UV = fws + FOFF_UV;

  const __bf16* whhg = bws + OFF_WHH;
  const __bf16* wihg = bws + OFF_WIH;
  __bf16* hbuf = bws + OFF_HBUF;
  __bf16* emgc = bws + OFF_EMG;
  __bf16* emgp = bws + OFF_EMGP;
  __bf16* fbv = bws + OFF_FBV;
  __bf16* chfb = bws + OFF_CHFB;
  __bf16* afffb = bws + OFF_AFFFB;
  const __bf16* affpre = bws + OFF_AFF;
  const __bf16* w1ps = bws + OFF_W1PS;
  const __bf16* w2t = bws + OFF_W2TAP;
  const __bf16* wrdo = bws + OFF_WRDO;
  const __bf16* wfb = bws + OFF_WFB;
  const int eon = *wees;

  // load W_hh shard (rows {j, 512+j, 1024+j : j in wg*16..wg*16+15}) into LDS
  for (int i = tid; i < 48 * HID / 8; i += WGTH) {
    int lr = i / (HID / 8), off = i % (HID / 8);
    int grow = (lr / 16) * HID + wg * 16 + (lr % 16);
    ((uint4*)s_whh)[i] = ((const uint4*)whhg)[(size_t)grow * (HID / 8) + off];
  }
  for (int i = tid; i < B_ * HID / 8; i += WGTH)
    ((uint4*)s_h)[i] = make_uint4(0u, 0u, 0u, 0u);
  if (wg == 0) {
    for (int i = tid; i < B_ * CEMG; i += WGTH) {
      emgc[i] = (__bf16)0.f;
      emgp[i] = (__bf16)0.f;
    }
  }
  gbar(bar, gen);
  int par = 0;

  for (int bin = 0; bin < NB; bin++) {
    const bool fbbin = (bin >= T0);
    if (fbbin) {
      // -- P0a: V_j = emg_prev @ PS_j^T, U_j = emg @ PS_j^T  (folded conv1)
      for (int t5 = wg * 8 + wave; t5 < 640; t5 += NWG * 8) {
        int which = t5 / 320;
        int r = t5 % 320;
        int j = r / 64;
        int r2 = r % 64;
        int mT = r2 / 4, nT = r2 % 4;
        const __bf16* emgX = which ? emgc : emgp;
        const __bf16* Aj = w1ps + (size_t)j * CCH * CKIN;
        v8f acc = vzero8();
        const __bf16* colp = emgX + (size_t)(nT * 16 + nn) * CEMG;
#pragma unroll
        for (int kk = 0; kk < CEMG; kk += 32)
          acc = wmma_bf16(load_a16(Aj, CKIN, mT * 16, kk, lane),
                          load_b16(colp, kk, lane), acc);
        int b = nT * 16 + nn;
        float* dst = UV + (((size_t)which * 5 + j) * B_ + b) * CCH;
#pragma unroll
        for (int i2 = 0; i2 < 8; i2++) dst[mT * 16 + i2 + 8 * lg] = acc[i2];
      }
      gbar(bar, gen);
      // -- P0b: assemble ch_fb = relu(V[5-c2]-V[c0]+U[5]-U[5-c2]+b1)+ees
      for (int idx = wg * WGTH + tid; idx < B_ * TB * CCH; idx += NWG * WGTH) {
        int b = idx / (TB * CCH);
        int r = idx % (TB * CCH);
        int tt = r / CCH, c = r % CCH;
        int c0 = 4 - tt; if (c0 < 0) c0 = 0;
        int c2 = tt - 24; if (c2 < 0) c2 = 0; if (c2 > 5) c2 = 5;
        int a1 = 5 - c2;
        const float* Vb = UV;
        const float* Ub = UV + (size_t)5 * B_ * CCH;
        float v = Ub[((size_t)4 * B_ + b) * CCH + c];  // U_5
        if (a1 > 0) {
          v += Vb[(((size_t)(a1 - 1)) * B_ + b) * CCH + c];
          v -= Ub[(((size_t)(a1 - 1)) * B_ + b) * CCH + c];
        }
        if (c0 > 0) v -= Vb[(((size_t)(c0 - 1)) * B_ + b) * CCH + c];
        v += b1[c];
        v = v > 0.f ? v : 0.f;
        if (eon) v += ees[((size_t)b * CCH + c) * LTOT + bin * TI + tt];
        chfb[((size_t)b * TP + 4 + tt) * CCH + c] = (__bf16)v;
      }
      gbar(bar, gen);
      // -- P0c: conv2 on feedback channels -> aff_fb (4 N-tiles per wave)
      for (int job = wg * 8 + wave; job < 32 * 50; job += NWG * 8) {
        int mT = job / 50, nb = job % 50;
        int bj[4], ttj[4];
        const __bf16* cb[4];
#pragma unroll
        for (int j = 0; j < 4; j++) {
          int col = (nb * 4 + j) * 16 + nn;
          bj[j] = col / TB;
          ttj[j] = col % TB;
          cb[j] = chfb + ((size_t)bj[j] * TP + ttj[j]) * CCH;
        }
        v8f acc[4];
#pragma unroll
        for (int j = 0; j < 4; j++) acc[j] = vzero8();
        for (int k = 0; k < KE; k++) {
          const __bf16* Ak = w2t + (size_t)k * CAFF * CCH;
#pragma unroll
          for (int kk = 0; kk < CCH; kk += 32) {
            v16bf av = load_a16(Ak, CCH, mT * 16, kk, lane);
            v16bf bv[4];
#pragma unroll
            for (int j = 0; j < 4; j++) bv[j] = load_b16(cb[j] + k * CCH, kk, lane);
#pragma unroll
            for (int j = 0; j < 4; j++) acc[j] = wmma_bf16(av, bv[j], acc[j]);
          }
        }
#pragma unroll
        for (int j = 0; j < 4; j++) {
          __bf16* dst = afffb + ((size_t)bj[j] * TB + ttj[j]) * CAFF;
#pragma unroll
          for (int i2 = 0; i2 < 8; i2++) {
            int co = mT * 16 + i2 + 8 * lg;
            float v = acc[j][i2] + b2[co];
            dst[co] = (__bf16)(v > 0.f ? v : 0.f);
          }
        }
      }
      gbar(bar, gen);
    }
    // -- P1: gifb = W_ih(rows of this WG) @ fb  (skip at bin 0)
    if (bin > 0) {
      for (int t7 = wave; t7 < 12; t7 += 8) {
        int mm = t7 >> 2, nT = t7 & 3;
        v8f acc = vzero8();
        const __bf16* colp = fbv + (size_t)(nT * 16 + nn) * CAFF;
        for (int kk = 0; kk < CAFF; kk += 32)
          acc = wmma_bf16(load_a16(wihg, HID, mm * HID + wg * 16, kk, lane),
                          load_b16(colp, kk, lane), acc);
#pragma unroll
        for (int i2 = 0; i2 < 8; i2++)
          s_gifb[(mm * 16 + i2 + 8 * lg) * B_ + nT * 16 + nn] = acc[i2];
      }
    } else {
      for (int i = tid; i < 48 * B_; i += WGTH) s_gifb[i] = 0.f;
    }
    __syncthreads();
    // -- P2: gi[t][b][48rows] = W_ih(rows) @ aff + b_ih + gifb
    //        register-blocked 4 N-tiles per wave, A reused, B hoisted
    const __bf16* affsrc =
        fbbin ? (const __bf16*)afffb : affpre + (size_t)bin * B_ * TB * CAFF;
    for (int job = wave; job < 3 * 50; job += 8) {
      int mm = job / 50, nb = job % 50;
      int colj[4];
      const __bf16* cb[4];
#pragma unroll
      for (int j = 0; j < 4; j++) {
        colj[j] = (nb * 4 + j) * 16 + nn;
        cb[j] = affsrc + (size_t)colj[j] * CAFF;
      }
      v8f acc[4];
#pragma unroll
      for (int j = 0; j < 4; j++) acc[j] = vzero8();
#pragma unroll 2
      for (int kk = 0; kk < CAFF; kk += 32) {
        v16bf av = load_a16(wihg, HID, mm * HID + wg * 16, kk, lane);
        v16bf bv[4];
#pragma unroll
        for (int j = 0; j < 4; j++) bv[j] = load_b16(cb[j], kk, lane);
#pragma unroll
        for (int j = 0; j < 4; j++) acc[j] = wmma_bf16(av, bv[j], acc[j]);
      }
#pragma unroll
      for (int j = 0; j < 4; j++) {
        int b = colj[j] / TB, tt = colj[j] % TB;
#pragma unroll
        for (int i2 = 0; i2 < 8; i2++) {
          int lr = mm * 16 + i2 + 8 * lg;
          int grow = mm * HID + wg * 16 + i2 + 8 * lg;
          giW[((size_t)tt * B_ + b) * 48 + lr] =
              acc[j][i2] + b_ih[grow] + s_gifb[lr * B_ + b];
        }
      }
    }
    __syncthreads();
    // -- P3: 50 sequential GRU steps (12 tiles spread over 8 waves)
    for (int s = 0; s < TB; s++) {
      for (int t9 = wave; t9 < 12; t9 += 8) {
        int mm = t9 >> 2, nT = t9 & 3;
        v8f acc = vzero8();
        const __bf16* colp = s_h + (size_t)(nT * 16 + nn) * HID;
        for (int kk = 0; kk < HID; kk += 32)
          acc = wmma_bf16(load_a16(s_whh, HID, mm * 16, kk, lane),
                          load_b16(colp, kk, lane), acc);
#pragma unroll
        for (int i2 = 0; i2 < 8; i2++) {
          int lr = mm * 16 + i2 + 8 * lg;
          s_gh[lr * B_ + nT * 16 + nn] =
              acc[i2] + b_hh[mm * HID + wg * 16 + i2 + 8 * lg];
        }
      }
      __syncthreads();
      for (int idx = tid; idx < 16 * B_; idx += WGTH) {
        int u = idx >> 6, b = idx & 63;
        const float* gis = giW + ((size_t)s * B_ + b) * 48;
        float r = sigm(gis[u] + s_gh[u * B_ + b]);
        float z = sigm(gis[16 + u] + s_gh[(16 + u) * B_ + b]);
        float n2 = tanhf(gis[32 + u] + r * s_gh[(32 + u) * B_ + b]);
        int hc = wg * 16 + u;
        float hold = (float)s_h[(size_t)b * HID + hc];
        float hnew = (1.f - z) * n2 + z * hold;
        hbuf[(size_t)par * B_ * HID + (size_t)b * HID + hc] = (__bf16)hnew;
      }
      gbar(bar, gen);
      {  // refresh full h from the exchange buffer
        const uint4* src = (const uint4*)(hbuf + (size_t)par * B_ * HID);
        uint4* dst = (uint4*)s_h;
        for (int i = tid; i < B_ * HID / 8; i += WGTH) dst[i] = src[i];
      }
      __syncthreads();
      par ^= 1;
    }
    // -- P4: EMG readout (WG0): emg = h @ Wrdo_last^T + b_rdo
    if (wg == 0) {
      for (int i = tid; i < B_ * CEMG; i += WGTH)
        emgp[i] = (bin > 2) ? emgc[i] : emgp[i];
      __syncthreads();
      for (int t10 = wave; t10 < 16; t10 += 8) {
        int mT = t10 >> 2, nT = t10 & 3;
        v8f acc = vzero8();
        const __bf16* colp = s_h + (size_t)(nT * 16 + nn) * HID;
        for (int kk = 0; kk < HID; kk += 32)
          acc = wmma_bf16(load_a16(wrdo, HID, mT * 16, kk, lane),
                          load_b16(colp, kk, lane), acc);
        int b = nT * 16 + nn;
#pragma unroll
        for (int i2 = 0; i2 < 8; i2++) {
          int e = mT * 16 + i2 + 8 * lg;
          float v = acc[i2] + b_rdo[e];
          out[((size_t)b * CEMG + e) * NB + bin] = v;
          emgc[(size_t)b * CEMG + e] = (__bf16)v;
        }
      }
    }
    gbar(bar, gen);
    // -- P5: feedback projection fb = emg @ W_fb^T + b_fb
    for (int t11 = wg * 8 + wave; t11 < 128; t11 += NWG * 8) {
      int mT = t11 >> 2, nT = t11 & 3;
      v8f acc = vzero8();
      const __bf16* colp = emgc + (size_t)(nT * 16 + nn) * CEMG;
#pragma unroll
      for (int kk = 0; kk < CEMG; kk += 32)
        acc = wmma_bf16(load_a16(wfb, CEMG, mT * 16, kk, lane),
                        load_b16(colp, kk, lane), acc);
      int b = nT * 16 + nn;
#pragma unroll
      for (int i2 = 0; i2 < 8; i2++) {
        int a = mT * 16 + i2 + 8 * lg;
        fbv[(size_t)b * CAFF + a] = (__bf16)(acc[i2] + b_fb[a]);
      }
    }
    gbar(bar, gen);
  }
}

// ---------------------------------------------------------------------------
extern "C" void kernel_launch(void* const* d_in, const int* in_sizes, int n_in,
                              void* d_out, int out_size, void* d_ws, size_t ws_size,
                              hipStream_t stream) {
  (void)in_sizes; (void)n_in; (void)out_size; (void)ws_size;
  const float* kin = (const float*)d_in[0];
  const float* ees = (const float*)d_in[1];
  const float* We1 = (const float*)d_in[2];
  const float* be1 = (const float*)d_in[3];
  const float* We2 = (const float*)d_in[4];
  const float* be2 = (const float*)d_in[5];
  const float* Wih = (const float*)d_in[6];
  const float* Whh = (const float*)d_in[7];
  const float* bih = (const float*)d_in[8];
  const float* bhh = (const float*)d_in[9];
  const float* Wrdo = (const float*)d_in[10];
  const float* brdo = (const float*)d_in[11];
  const float* Wfb = (const float*)d_in[12];
  const float* bfb = (const float*)d_in[13];
  const int* wees = (const int*)d_in[14];
  float* out = (float*)d_out;

  __bf16* bws = (__bf16*)d_ws;
  size_t bf16_bytes = ((BF16_ELEMS * 2 + 255) / 256) * 256;  // ~120 MB
  float* fws = (float*)((char*)d_ws + bf16_bytes);           // ~20 MB more

  // reset the global-barrier counter (monotonic within one launch)
  hipMemsetAsync((void*)(fws + FOFF_BAR), 0, 64 * sizeof(float), stream);

  // K0: weight conversion / padded-window construction
  k_prep<<<32256, 256, 0, stream>>>(We1, We2, Wih, Whh, Wrdo, Wfb, kin, bws);
  // K1/K2: parallel afferent precompute for bins 0..19
  k_conv1<<<(T0 * 16 * 50) / 8, 256, 0, stream>>>(bws, be1, ees, wees);
  k_conv2<<<(T0 * 32 * 50) / 8, 256, 0, stream>>>(bws, be2);
  // K3: persistent GRU engine (32 WGs, ~136 KB LDS each)
  size_t smem = (size_t)(48 * HID + B_ * HID) * sizeof(__bf16) +
                (size_t)(2 * 48 * B_) * sizeof(float);
  k_recur<<<NWG, WGTH, smem, stream>>>(bws, fws, out, be1, be2, bih, bhh, brdo,
                                       bfb, ees, wees);
}
